// GraphTransformerModule_14508399526690
// MI455X (gfx1250) — matrix operand order repaired and checked by reading, not compiled, not run on hardware
//
#include <hip/hip_runtime.h>

typedef __attribute__((ext_vector_type(16))) _Float16 v16h;
typedef __attribute__((ext_vector_type(8)))  _Float16 v8h;
typedef __attribute__((ext_vector_type(8)))  float    v8f;

static constexpr int Dm = 128;   // model dim
static constexpr int NH = 8;     // heads
static constexpr int FF = 256;   // FFN hidden dim
static constexpr int SA_LD = Dm + 8;   // padded LDS stride (halves) for A operand
static constexpr int SH_LD = FF + 8;   // padded LDS stride (halves) for FFN hidden

// ---------------- WMMA helpers ----------------

// A operand (16x32 f16 tile) from LDS, per CDNA5 16-bit A layout:
// lanes 0-15 : row m=lane,   halves[0..7]=K[k0..k0+7],   halves[8..15]=K[k0+16..k0+23]
// lanes 16-31: row m=lane-16,halves[0..7]=K[k0+8..k0+15],halves[8..15]=K[k0+24..k0+31]
__device__ inline v16h lds_a_frag(const _Float16* base, int lda, int k0) {
  int lane = threadIdx.x & 31;
  int m = lane & 15;
  int koff = (lane & 16) ? 8 : 0;
  const _Float16* p = base + m * lda + k0 + koff;
  v8h lo = *(const v8h*)p;
  v8h hi = *(const v8h*)(p + 16);
  v16h a;
#pragma unroll
  for (int j = 0; j < 8; ++j) { a[j] = lo[j]; a[j + 8] = hi[j]; }
  return a;
}

// B operand: weights pre-swizzled so each lane reads 16 contiguous halves (32B).
__device__ inline v16h glb_b_frag(const _Float16* wmat, int ktiles, int nt, int kt) {
  int lane = threadIdx.x & 31;
  const _Float16* p = wmat + ((size_t)(nt * ktiles + kt) << 9) + (lane << 4);
  return *(const v16h*)p;
}

template <int KT>
__device__ inline v8f wmma_tile(const _Float16* aLds, int lda,
                                const _Float16* wmat, int nt, v8f c) {
#pragma unroll
  for (int kt = 0; kt < KT; ++kt) {
    v16h a = lds_a_frag(aLds, lda, kt * 32);
    v16h b = glb_b_frag(wmat, KT, nt, kt);
    c = __builtin_amdgcn_wmma_f32_16x16x32_f16(false, a, false, b, (short)0, c,
                                               false, false);
  }
  return c;
}

// ---------------- LayerNorm over 16 rows x 128 cols ----------------
__device__ inline void ln_16xD(const float* x, int xld, _Float16* out, int old_,
                               const float* __restrict__ gamma,
                               const float* __restrict__ beta) {
  int tid = threadIdx.x;
  int r = tid >> 4;
  int cc = (tid & 15) * 8;
  float v[8];
  float s = 0.f, s2 = 0.f;
#pragma unroll
  for (int j = 0; j < 8; ++j) {
    v[j] = x[r * xld + cc + j];
    s += v[j];
    s2 += v[j] * v[j];
  }
#pragma unroll
  for (int off = 8; off; off >>= 1) {
    s += __shfl_xor(s, off, 16);
    s2 += __shfl_xor(s2, off, 16);
  }
  float mu = s * (1.f / 128.f);
  float var = s2 * (1.f / 128.f) - mu * mu;
  float rs = rsqrtf(var + 1e-5f);
#pragma unroll
  for (int j = 0; j < 8; ++j) {
    float y = (v[j] - mu) * rs * gamma[cc + j] + beta[cc + j];
    out[r * old_ + cc + j] = (_Float16)y;
  }
}

// ---------------- utility kernels ----------------

__global__ __launch_bounds__(256) void zero_kernel(float* __restrict__ p, size_t n) {
  size_t i = (size_t)blockIdx.x * 256 + threadIdx.x;
  if (i < n) p[i] = 0.f;
}

// Convert f32 row-major weight [Krows x NC] into f16 WMMA-B fragment order.
__global__ __launch_bounds__(256) void convert_swizzle(const float* __restrict__ W,
                                                       _Float16* __restrict__ out,
                                                       int Krows, int NC) {
  int i = blockIdx.x * 256 + threadIdx.x;
  if (i >= Krows * NC) return;
  int j = i & 15;
  int lane = (i >> 4) & 31;
  int tile = i >> 9;
  int ktiles = Krows >> 5;
  int kt = tile % ktiles;
  int nt = tile / ktiles;
  int k = kt * 32 + ((lane & 16) ? 8 : 0) + (j & 7) + ((j & 8) ? 16 : 0);
  int n = nt * 16 + (lane & 15);
  out[i] = (_Float16)W[k * NC + n];
}

// ---------------- node kernel A: LN1 + Q/K/V projections (f16 out) ----------------

__global__ __launch_bounds__(256) void node_qkv_kernel(
    const float* __restrict__ nodeF, const float* __restrict__ g1,
    const float* __restrict__ b1, const _Float16* __restrict__ wq,
    const _Float16* __restrict__ wk, const _Float16* __restrict__ wv,
    _Float16* __restrict__ Qo, _Float16* __restrict__ Ko,
    _Float16* __restrict__ Vo, int Ntot) {
  __shared__ float sX[16 * Dm];
  __shared__ _Float16 sA[16 * SA_LD];
  int tid = threadIdx.x;
  int n0 = blockIdx.x * 16;
  const bool full = (n0 + 16 <= Ntot);  // block-uniform
  int r = tid >> 4, cc = (tid & 15) * 8;
  int row = n0 + r;
  if (row >= Ntot) row = Ntot - 1;
#pragma unroll
  for (int j = 0; j < 8; ++j) sX[r * Dm + cc + j] = nodeF[(size_t)row * Dm + cc + j];
  __syncthreads();
  ln_16xD(sX, Dm, sA, SA_LD, g1, b1);
  __syncthreads();

  int wave = tid >> 5, lane = tid & 31;
  int colb = wave * 16 + (lane & 15);
  int rb = (lane & 16) ? 8 : 0;
  const _Float16* mats[3] = {wq, wk, wv};
  _Float16* outs[3] = {Qo, Ko, Vo};
#pragma unroll
  for (int m = 0; m < 3; ++m) {
    v8f c = {};
    c = wmma_tile<4>(sA, SA_LD, mats[m], wave, c);
    if (full) {
#pragma unroll
      for (int v = 0; v < 8; ++v)
        outs[m][(size_t)(n0 + rb + v) * Dm + colb] = (_Float16)c[v];
    } else {
#pragma unroll
      for (int v = 0; v < 8; ++v) {
        int rr = n0 + rb + v;
        if (rr < Ntot) outs[m][(size_t)rr * Dm + colb] = (_Float16)c[v];
      }
    }
  }
}

// ---------------- edge kernel: fully fused edge pipeline ----------------

__global__ __launch_bounds__(256) void edge_kernel(
    const float* __restrict__ edgeF, const int* __restrict__ eidx,
    const _Float16* __restrict__ Qg, const _Float16* __restrict__ Kg,
    const _Float16* __restrict__ Vg, const _Float16* __restrict__ wWe,
    const _Float16* __restrict__ wWoe, const _Float16* __restrict__ wW1,
    const _Float16* __restrict__ wW2, const float* __restrict__ boe,
    const float* __restrict__ g1, const float* __restrict__ b1,
    const float* __restrict__ g2, const float* __restrict__ b2,
    float* __restrict__ wVacc, float* __restrict__ zacc,
    float* __restrict__ eOut, int Etot) {
  __shared__ float sF[16 * Dm];        // e_in, later e2
  __shared__ float sP[16 * Dm];        // PE (f32)
  __shared__ _Float16 sA[16 * SA_LD];  // f16 A operand (LN out / scores)
  __shared__ _Float16 sH[16 * SH_LD];  // FFN hidden
  __shared__ int sSrc[16], sDst[16];

  int tid = threadIdx.x;
  int e0 = blockIdx.x * 16;
  const bool full = (e0 + 16 <= Etot);  // block-uniform
  __builtin_prefetch(wWe, 0, 3);
  __builtin_prefetch(wWoe, 0, 3);
  __builtin_prefetch(wW1, 0, 3);
  __builtin_prefetch(wW2, 0, 3);
  if (tid < 16) {
    int e = e0 + tid;
    if (e >= Etot) e = Etot - 1;
    sSrc[tid] = eidx[e];
    sDst[tid] = eidx[Etot + e];
  }
  int r = tid >> 4, cc = (tid & 15) * 8;
  int erow = e0 + r;
  if (erow >= Etot) erow = Etot - 1;
#pragma unroll
  for (int j = 0; j < 8; ++j) sF[r * Dm + cc + j] = edgeF[(size_t)erow * Dm + cc + j];
  __syncthreads();
  ln_16xD(sF, Dm, sA, SA_LD, g1, b1);
  __syncthreads();

  int wave = tid >> 5, lane = tid & 31;
  int colb = wave * 16 + (lane & 15);
  int rb = (lane & 16) ? 8 : 0;

  // PE = LN(e) @ We  -> sP
  {
    v8f c = {};
    c = wmma_tile<4>(sA, SA_LD, wWe, wave, c);
#pragma unroll
    for (int v = 0; v < 8; ++v) sP[(rb + v) * Dm + colb] = c[v];
  }
  __syncthreads();

  // Per-edge attention: score = clip(K[src]*Q[dst]/4) * PE; w = exp(clip(head-sum))
  {
    int src = sSrc[r], dst = sDst[r];
    v8h kvec = *(const v8h*)(Kg + (size_t)src * Dm + cc);
    v8h qvec = *(const v8h*)(Qg + (size_t)dst * Dm + cc);
    float part = 0.f;
#pragma unroll
    for (int j = 0; j < 8; ++j) {
      float kq = (float)kvec[j] * (float)qvec[j] * 0.25f;
      kq = fminf(fmaxf(kq, -5.f), 5.f);
      float sc = kq * sP[r * Dm + cc + j];
      part += sc;
      sA[r * SA_LD + cc + j] = (_Float16)sc;  // e_attn (A operand for Wo_e)
    }
    float hs = part + __shfl_xor(part, 1, 32);  // pair covers one 16-wide head
    hs = fminf(fmaxf(hs, -5.f), 5.f);
    float w = expf(hs);
    int h = (tid & 15) >> 1;
    if (full) {
      v8h vvec = *(const v8h*)(Vg + (size_t)src * Dm + cc);
      if ((tid & 1) == 0) atomicAdd(&zacc[(size_t)dst * NH + h], w);
#pragma unroll
      for (int j = 0; j < 8; ++j)
        atomicAdd(&wVacc[(size_t)dst * Dm + cc + j], (float)vvec[j] * w);
    } else if ((e0 + r) < Etot) {
      v8h vvec = *(const v8h*)(Vg + (size_t)src * Dm + cc);
      if ((tid & 1) == 0) atomicAdd(&zacc[(size_t)dst * NH + h], w);
#pragma unroll
      for (int j = 0; j < 8; ++j)
        atomicAdd(&wVacc[(size_t)dst * Dm + cc + j], (float)vvec[j] * w);
    }
  }
  __syncthreads();

  // e2 = e_in + e_attn @ Wo_e + bo_e  (into sF)
  {
    v8f c = {};
    c = wmma_tile<4>(sA, SA_LD, wWoe, wave, c);
#pragma unroll
    for (int v = 0; v < 8; ++v) {
      int m = rb + v;
      sF[m * Dm + colb] = sF[m * Dm + colb] + c[v] + boe[colb];
    }
  }
  __syncthreads();
  ln_16xD(sF, Dm, sA, SA_LD, g2, b2);
  __syncthreads();

  // FFN1 + SiLU -> sH  (256 cols: each wave does 2 tiles)
#pragma unroll
  for (int t = 0; t < 2; ++t) {
    int nt = wave + t * 8;
    v8f c = {};
    c = wmma_tile<4>(sA, SA_LD, wW1, nt, c);
#pragma unroll
    for (int v = 0; v < 8; ++v) {
      float x = c[v];
      sH[(rb + v) * SH_LD + nt * 16 + (lane & 15)] =
          (_Float16)(x / (1.f + expf(-x)));
    }
  }
  __syncthreads();

  // e_out = e2 + hidden @ W2_e
  {
    v8f c = {};
    c = wmma_tile<8>(sH, SH_LD, wW2, wave, c);
    if (full) {
#pragma unroll
      for (int v = 0; v < 8; ++v) {
        int m = rb + v;
        eOut[(size_t)(e0 + m) * Dm + colb] = sF[m * Dm + colb] + c[v];
      }
    } else {
#pragma unroll
      for (int v = 0; v < 8; ++v) {
        int m = rb + v;
        int er = e0 + m;
        if (er < Etot) eOut[(size_t)er * Dm + colb] = sF[m * Dm + colb] + c[v];
      }
    }
  }
}

// ---------------- node kernel B: aggregate + output proj + FFN ----------------

__global__ __launch_bounds__(256) void node_out_kernel(
    const float* __restrict__ nodeF, const float* __restrict__ wVacc,
    const float* __restrict__ zacc, const _Float16* __restrict__ wWon,
    const float* __restrict__ bon, const float* __restrict__ g2,
    const float* __restrict__ b2, const _Float16* __restrict__ wW1,
    const _Float16* __restrict__ wW2, float* __restrict__ hOut, int Ntot) {
  __shared__ float sF[16 * Dm];
  __shared__ _Float16 sA[16 * SA_LD];
  __shared__ _Float16 sH[16 * SH_LD];
  int tid = threadIdx.x;
  int n0 = blockIdx.x * 16;
  const bool full = (n0 + 16 <= Ntot);  // block-uniform
  int r = tid >> 4, cc = (tid & 15) * 8;
  int row = n0 + r;
  if (row >= Ntot) row = Ntot - 1;
  float zz = zacc[(size_t)row * NH + ((tid & 15) >> 1)];
  float inv = 1.f / (zz + 1e-6f);
#pragma unroll
  for (int j = 0; j < 8; ++j) {
    sF[r * Dm + cc + j] = nodeF[(size_t)row * Dm + cc + j];
    sA[r * SA_LD + cc + j] = (_Float16)(wVacc[(size_t)row * Dm + cc + j] * inv);
  }
  __syncthreads();

  int wave = tid >> 5, lane = tid & 31;
  int colb = wave * 16 + (lane & 15);
  int rb = (lane & 16) ? 8 : 0;

  // h2 = h_in + h_attn @ Wo_n + bo_n
  {
    v8f c = {};
    c = wmma_tile<4>(sA, SA_LD, wWon, wave, c);
#pragma unroll
    for (int v = 0; v < 8; ++v) {
      int m = rb + v;
      sF[m * Dm + colb] = sF[m * Dm + colb] + c[v] + bon[colb];
    }
  }
  __syncthreads();
  ln_16xD(sF, Dm, sA, SA_LD, g2, b2);
  __syncthreads();

#pragma unroll
  for (int t = 0; t < 2; ++t) {
    int nt = wave + t * 8;
    v8f c = {};
    c = wmma_tile<4>(sA, SA_LD, wW1, nt, c);
#pragma unroll
    for (int v = 0; v < 8; ++v) {
      float x = c[v];
      sH[(rb + v) * SH_LD + nt * 16 + (lane & 15)] =
          (_Float16)(x / (1.f + expf(-x)));
    }
  }
  __syncthreads();
  {
    v8f c = {};
    c = wmma_tile<8>(sH, SH_LD, wW2, wave, c);
    if (full) {
#pragma unroll
      for (int v = 0; v < 8; ++v) {
        int m = rb + v;
        hOut[(size_t)(n0 + m) * Dm + colb] = sF[m * Dm + colb] + c[v];
      }
    } else {
#pragma unroll
      for (int v = 0; v < 8; ++v) {
        int m = rb + v;
        int orow = n0 + m;
        if (orow < Ntot) hOut[(size_t)orow * Dm + colb] = sF[m * Dm + colb] + c[v];
      }
    }
  }
}

// ---------------- host launcher ----------------

extern "C" void kernel_launch(void* const* d_in, const int* in_sizes, int n_in,
                              void* d_out, int out_size, void* d_ws, size_t ws_size,
                              hipStream_t stream) {
  const float* nodeF = (const float*)d_in[0];
  const float* edgeF = (const float*)d_in[1];
  const int* eidx = (const int*)d_in[2];
  const float* Wq = (const float*)d_in[3];
  const float* Wk = (const float*)d_in[4];
  const float* Wv = (const float*)d_in[5];
  const float* We = (const float*)d_in[6];
  const float* Won = (const float*)d_in[7];
  const float* bon = (const float*)d_in[8];
  const float* Woe = (const float*)d_in[9];
  const float* boe = (const float*)d_in[10];
  const float* W1n = (const float*)d_in[11];
  const float* W2n = (const float*)d_in[12];
  const float* W1e = (const float*)d_in[13];
  const float* W2e = (const float*)d_in[14];
  const float* g1n = (const float*)d_in[15];
  const float* b1n = (const float*)d_in[16];
  const float* g1e = (const float*)d_in[17];
  const float* b1e = (const float*)d_in[18];
  const float* g2n = (const float*)d_in[19];
  const float* b2n = (const float*)d_in[20];
  const float* g2e = (const float*)d_in[21];
  const float* b2e = (const float*)d_in[22];

  const int Ntot = in_sizes[0] / Dm;
  const int Etot = in_sizes[1] / Dm;

  // workspace layout: Q | K | V (f16) | wV | z (f32) | f16 weights (~66 MB total)
  _Float16* Qb = (_Float16*)d_ws;
  _Float16* Kb = Qb + (size_t)Ntot * Dm;
  _Float16* Vb = Kb + (size_t)Ntot * Dm;
  float* wVb = (float*)(Vb + (size_t)Ntot * Dm);  // 768*N bytes before -> 4B aligned
  float* zb = wVb + (size_t)Ntot * Dm;
  size_t half_off = (size_t)(zb + (size_t)Ntot * NH);
  half_off = (half_off + 63) & ~(size_t)63;  // 64B align for v16h loads
  _Float16* hWq = (_Float16*)half_off;
  _Float16* hWk = hWq + 16384;
  _Float16* hWv = hWk + 16384;
  _Float16* hWe = hWv + 16384;
  _Float16* hWon = hWe + 16384;
  _Float16* hWoe = hWon + 16384;
  _Float16* hW1n = hWoe + 16384;
  _Float16* hW2n = hW1n + 32768;
  _Float16* hW1e = hW2n + 32768;
  _Float16* hW2e = hW1e + 32768;

  dim3 blk(256);

  // 1) weight convert + swizzle to WMMA-B order
  convert_swizzle<<<64, blk, 0, stream>>>(Wq, hWq, 128, 128);
  convert_swizzle<<<64, blk, 0, stream>>>(Wk, hWk, 128, 128);
  convert_swizzle<<<64, blk, 0, stream>>>(Wv, hWv, 128, 128);
  convert_swizzle<<<64, blk, 0, stream>>>(We, hWe, 128, 128);
  convert_swizzle<<<64, blk, 0, stream>>>(Won, hWon, 128, 128);
  convert_swizzle<<<64, blk, 0, stream>>>(Woe, hWoe, 128, 128);
  convert_swizzle<<<128, blk, 0, stream>>>(W1n, hW1n, 128, 256);
  convert_swizzle<<<128, blk, 0, stream>>>(W2n, hW2n, 256, 128);
  convert_swizzle<<<128, blk, 0, stream>>>(W1e, hW1e, 128, 256);
  convert_swizzle<<<128, blk, 0, stream>>>(W2e, hW2e, 256, 128);

  // 2) zero segment-sum accumulators (wV and z are contiguous)
  size_t zn = (size_t)Ntot * Dm + (size_t)Ntot * NH;
  zero_kernel<<<(unsigned)((zn + 255) / 256), blk, 0, stream>>>(wVb, zn);

  // 3) node LN1 + QKV
  node_qkv_kernel<<<(Ntot + 15) / 16, blk, 0, stream>>>(nodeF, g1n, b1n, hWq, hWk,
                                                        hWv, Qb, Kb, Vb, Ntot);

  // 4) fused edge pipeline (writes e_out section of d_out)
  float* hOut = (float*)d_out;
  float* eOut = hOut + (size_t)Ntot * Dm;
  edge_kernel<<<(Etot + 15) / 16, blk, 0, stream>>>(
      edgeF, eidx, Qb, Kb, Vb, hWe, hWoe, hW1e, hW2e, boe, g1e, b1e, g2e, b2e,
      wVb, zb, eOut, Etot);

  // 5) node aggregation + output projection + FFN (writes h_out section)
  node_out_kernel<<<(Ntot + 15) / 16, blk, 0, stream>>>(
      nodeF, wVb, zb, hWon, bon, g2n, b2n, hW1n, hW2n, hOut, Ntot);
}
